// AutoregressiveStep_53412213293363
// MI455X (gfx1250) — matrix operand order repaired and checked by reading, not compiled, not run on hardware
//
#include <hip/hip_runtime.h>

// ---------------------------------------------------------------------------
// Stacked LSTM step (3 layers) + projection for MI455X (gfx1250, wave32).
// Weight-bandwidth-bound: fp32 weights converted to bf16 in-register and fed
// through v_wmma_f32_16x16x32_bf16. Gate fusion keeps Z[256,4096] in VGPRs.
// Round 2: LDS double buffering (stage k+1 overlaps WMMA of k, one barrier
// per K-step) + packed b32 LDS stores for the B (weight) tile.
// ---------------------------------------------------------------------------

typedef __attribute__((ext_vector_type(16))) __bf16 v16bf;
typedef __attribute__((ext_vector_type(2)))  __bf16 v2bf;
typedef __attribute__((ext_vector_type(8)))  float  v8f;

#define B_SZ 256
#define HID  1024
#define IN_SZ 512
#define VOC  512

#define MTILES 2              // 16-row sub-tiles per block -> 32 rows
#define ROWS   (MTILES * 16)
#define NCOLS  64             // H-columns per block (4 waves x 16)
#define KSTEP  32
#define NTHREADS 128
#define ABUF (MTILES * 512)   // bf16 elems per A buffer

__device__ __forceinline__ float sigmoidf_(float x) {
  return 1.0f / (1.0f + __expf(-x));
}

// ---------------------------------------------------------------------------
// Stage one K-tile (KSTEP deep) of A and of NG gate-tiles of B into LDS in
// *fragment-ready* order (so consumers do contiguous 32B ds_load_b128).
//
// A-fragment order staged here (16-bit A 16x32): lane L (m = L&15, half = L>>4):
//   e 0..15 -> K = 16*half + e   (contiguous, same order as B)
// B-fragment order staged here (16-bit B 32x16): lane L (n = L&15, kh = L>>4):
//   e 0..15 -> K = 16*kh + e   (K-consecutive per lane -> pack (K,K+1) pairs
//   from two adjacent weight rows into single b32 LDS stores)
// ---------------------------------------------------------------------------
template <int NG>
__device__ __forceinline__ void stage_tile(
    const float* __restrict__ A, int lda, int rowBase,
    const float* __restrict__ Wsrc, int ldw, int colBase, int gateStride,
    int k0, bool prefetch_next, __bf16* Af, __bf16* Bf)
{
  const int tid = threadIdx.x;

  // ---- A tile: ROWS x 32 fp32 -> bf16 ------------------------------------
  #pragma unroll
  for (int rep = 0; rep < MTILES; ++rep) {
    int t  = rep * NTHREADS + tid;           // 0 .. MTILES*128-1
    int m  = t >> 3;                         // 0..ROWS-1
    int k4 = (t & 7) << 2;                   // 0,4,..,28
    float4 v = *(const float4*)(A + (size_t)(rowBase + m) * lda + k0 + k4);
    int sub = m >> 4, mm = m & 15;
    float vv[4] = {v.x, v.y, v.z, v.w};
    #pragma unroll
    for (int q = 0; q < 4; ++q) {
      int kk   = k4 + q;
      int half = kk >> 4;
      int e    = kk & 15;
      Af[sub * 512 + ((half << 4) + mm) * 16 + e] = (__bf16)vv[q];
    }
  }

  // ---- B tiles: per gate, 32(K) x 64(N) fp32 -> bf16, packed b32 stores ---
  #pragma unroll
  for (int g = 0; g < NG; ++g) {
    const float* src = Wsrc + (size_t)k0 * ldw + colBase + g * gateStride;
    #pragma unroll
    for (int rep = 0; rep < 2; ++rep) {
      int idx = rep * NTHREADS + tid;        // 0..255
      int kp  = idx >> 4;                    // K-pair index 0..15
      int n   = (idx & 15) << 2;             // 0..60
      int kk  = kp << 1;                     // even K
      float4 v0 = *(const float4*)(src + (size_t)kk * ldw + n);
      float4 v1 = *(const float4*)(src + (size_t)(kk + 1) * ldw + n);
      if (prefetch_next)  // next K-tile of weights (global_prefetch_b8)
        __builtin_prefetch(src + (size_t)(kk + KSTEP) * ldw + n, 0, 3);
      int waveN    = n >> 4;
      int laneBase = (kp >> 3) << 4;         // K-half -> lane group
      int e        = kk & 15;                // even element index
      int nn       = n & 15;
      float a0[4] = {v0.x, v0.y, v0.z, v0.w};
      float a1[4] = {v1.x, v1.y, v1.z, v1.w};
      #pragma unroll
      for (int q = 0; q < 4; ++q) {
        __bf16* p = Bf + (((g * 4 + waveN) * 32 + laneBase + nn + q) << 4) + e;
        *(v2bf*)p = (v2bf){(__bf16)a0[q], (__bf16)a1[q]};  // ds_store_b32
      }
    }
  }
}

// ---------------------------------------------------------------------------
// Double-buffered GEMM phase: acc[g][s] += A[rows,Ktot] x W[Ktot, cols(g)].
// ---------------------------------------------------------------------------
template <int NG>
__device__ __forceinline__ void mma_phase(
    const float* __restrict__ A, int lda, int rowBase,
    const float* __restrict__ Wsrc, int ldw, int colBase, int gateStride,
    int Ktot, __bf16* Af, __bf16* Bf, v8f (&acc)[NG][MTILES])
{
  const int w    = threadIdx.x >> 5;   // wave id 0..3
  const int lane = threadIdx.x & 31;
  const int BBUF = NG * 4 * 32 * 16;   // bf16 elems per B buffer

  stage_tile<NG>(A, lda, rowBase, Wsrc, ldw, colBase, gateStride,
                 /*k0=*/0, /*pf=*/(KSTEP < Ktot), Af, Bf);
  __syncthreads();

  int buf = 0;
  for (int k0 = 0; k0 < Ktot; k0 += KSTEP) {
    int nxt = buf ^ 1;
    if (k0 + KSTEP < Ktot)   // stage next tile into the other buffer
      stage_tile<NG>(A, lda, rowBase, Wsrc, ldw, colBase, gateStride,
                     k0 + KSTEP, /*pf=*/(k0 + 2 * KSTEP < Ktot),
                     Af + nxt * ABUF, Bf + nxt * BBUF);

    // ---- consume current buffer: WMMA ------------------------------------
    const __bf16* Ab = Af + buf * ABUF;
    const __bf16* Bb = Bf + buf * BBUF;
    v16bf afrag[MTILES];
    #pragma unroll
    for (int s = 0; s < MTILES; ++s)
      afrag[s] = *(const v16bf*)(Ab + s * 512 + lane * 16);
    #pragma unroll
    for (int g = 0; g < NG; ++g) {
      v16bf bfrag = *(const v16bf*)(Bb + (((g * 4 + w) * 32 + lane) << 4));
      #pragma unroll
      for (int s = 0; s < MTILES; ++s)
        acc[g][s] = __builtin_amdgcn_wmma_f32_16x16x32_bf16(
            false, afrag[s], false, bfrag, (short)0, acc[g][s], false, false);
    }
    __syncthreads();  // staging of nxt done AND everyone done reading buf
    buf = nxt;
  }
}

// ---------------------------------------------------------------------------
// Fused LSTM layer: Z = X@W + Hin@U + b ; gates -> (Hout, Cout)
// Grid: (B/32, H/64). Block: 128 threads (4 wave32).
// ---------------------------------------------------------------------------
__global__ void __launch_bounds__(NTHREADS)
lstm_layer_kernel(const float* __restrict__ X, int Kx,
                  const float* __restrict__ Hin,
                  const float* __restrict__ W,    // [Kx, 4H]
                  const float* __restrict__ U,    // [H, 4H]
                  const float* __restrict__ bias, // [4H]
                  const float* __restrict__ Cin,  // [B, H]
                  float* __restrict__ Hout,
                  float* __restrict__ Cout)
{
  __shared__ alignas(32) __bf16 Af[2 * ABUF];            // 4 KB
  __shared__ alignas(32) __bf16 Bf[2 * 4 * 4 * 32 * 16]; // 32 KB

  const int rowBase = blockIdx.x * ROWS;
  const int jBase   = blockIdx.y * NCOLS;

  v8f acc[4][MTILES];
  v8f zero = {};
  #pragma unroll
  for (int g = 0; g < 4; ++g)
    #pragma unroll
    for (int s = 0; s < MTILES; ++s) acc[g][s] = zero;

  // Phase 1: X @ W   (K = Kx, row stride Kx)
  mma_phase<4>(X, Kx, rowBase, W, 4 * HID, jBase, HID, Kx, Af, Bf, acc);
  // Phase 2: Hin @ U (K = H, row stride H)
  mma_phase<4>(Hin, HID, rowBase, U, 4 * HID, jBase, HID, HID, Af, Bf, acc);

  // Epilogue: bias + gates + cell update.  C/D layout: VGPR r of lane L holds
  // (M = r + 8*(L>>4), N = L&15).
  const int w    = threadIdx.x >> 5;
  const int lane = threadIdx.x & 31;
  const int col  = jBase + w * 16 + (lane & 15);
  const float bi = bias[col];
  const float bf = bias[HID + col];
  const float bg = bias[2 * HID + col];
  const float bo = bias[3 * HID + col];
  #pragma unroll
  for (int s = 0; s < MTILES; ++s) {
    const float* zi = (const float*)&acc[0][s];
    const float* zf = (const float*)&acc[1][s];
    const float* zg = (const float*)&acc[2][s];
    const float* zo = (const float*)&acc[3][s];
    #pragma unroll
    for (int r = 0; r < 8; ++r) {
      int row = rowBase + s * 16 + r + ((lane >> 4) << 3);
      size_t off = (size_t)row * HID + col;
      float ig = sigmoidf_(zi[r] + bi);
      float fg = sigmoidf_(zf[r] + bf);
      float gg = tanhf(zg[r] + bg);
      float og = sigmoidf_(zo[r] + bo);
      float cn = fg * Cin[off] + ig * gg;
      Cout[off] = cn;
      Hout[off] = og * tanhf(cn);
    }
  }
}

// ---------------------------------------------------------------------------
// Projection: logits = H2 @ Wp + bp.  Grid: (B/32, V/64).
// ---------------------------------------------------------------------------
__global__ void __launch_bounds__(NTHREADS)
proj_kernel(const float* __restrict__ H2,
            const float* __restrict__ Wp,   // [H, V]
            const float* __restrict__ bp,   // [V]
            float* __restrict__ Out)        // [B, V]
{
  __shared__ alignas(32) __bf16 Af[2 * ABUF];        // 4 KB
  __shared__ alignas(32) __bf16 Bf[2 * 4 * 32 * 16]; // 8 KB

  const int rowBase = blockIdx.x * ROWS;
  const int jBase   = blockIdx.y * NCOLS;

  v8f acc[1][MTILES];
  v8f zero = {};
  #pragma unroll
  for (int s = 0; s < MTILES; ++s) acc[0][s] = zero;

  mma_phase<1>(H2, HID, rowBase, Wp, VOC, jBase, 0, HID, Af, Bf, acc);

  const int w    = threadIdx.x >> 5;
  const int lane = threadIdx.x & 31;
  const int col  = jBase + w * 16 + (lane & 15);
  const float bb = bp[col];
  #pragma unroll
  for (int s = 0; s < MTILES; ++s) {
    const float* z = (const float*)&acc[0][s];
    #pragma unroll
    for (int r = 0; r < 8; ++r) {
      int row = rowBase + s * 16 + r + ((lane >> 4) << 3);
      Out[(size_t)row * VOC + col] = z[r] + bb;
    }
  }
}

// ---------------------------------------------------------------------------
extern "C" void kernel_launch(void* const* d_in, const int* in_sizes, int n_in,
                              void* d_out, int out_size, void* d_ws, size_t ws_size,
                              hipStream_t stream) {
  (void)in_sizes; (void)n_in; (void)out_size; (void)d_ws; (void)ws_size;
  const float* x  = (const float*)d_in[0];
  const float* h0 = (const float*)d_in[1];
  const float* c0 = (const float*)d_in[2];
  const float* h1 = (const float*)d_in[3];
  const float* c1 = (const float*)d_in[4];
  const float* h2 = (const float*)d_in[5];
  const float* c2 = (const float*)d_in[6];
  const float* W0 = (const float*)d_in[7];
  const float* U0 = (const float*)d_in[8];
  const float* b0 = (const float*)d_in[9];
  const float* W1 = (const float*)d_in[10];
  const float* U1 = (const float*)d_in[11];
  const float* b1 = (const float*)d_in[12];
  const float* W2 = (const float*)d_in[13];
  const float* U2 = (const float*)d_in[14];
  const float* b2 = (const float*)d_in[15];
  const float* Wp = (const float*)d_in[16];
  const float* bp = (const float*)d_in[17];

  float* out    = (float*)d_out;
  float* logits = out;                         // [B, V]
  float* h0n = logits + B_SZ * VOC;            // [B, H]
  float* c0n = h0n + B_SZ * HID;
  float* h1n = c0n + B_SZ * HID;
  float* c1n = h1n + B_SZ * HID;
  float* h2n = c1n + B_SZ * HID;
  float* c2n = h2n + B_SZ * HID;

  dim3 gridL(B_SZ / ROWS, HID / NCOLS);        // (8, 16)
  lstm_layer_kernel<<<gridL, NTHREADS, 0, stream>>>(x,   IN_SZ, h0, W0, U0, b0, c0, h0n, c0n);
  lstm_layer_kernel<<<gridL, NTHREADS, 0, stream>>>(h0n, HID,   h1, W1, U1, b1, c1, h1n, c1n);
  lstm_layer_kernel<<<gridL, NTHREADS, 0, stream>>>(h1n, HID,   h2, W2, U2, b2, c2, h2n, c2n);

  dim3 gridP(B_SZ / ROWS, VOC / NCOLS);        // (8, 8)
  proj_kernel<<<gridP, NTHREADS, 0, stream>>>(h2n, Wp, bp, logits);
}
